// MultiHeadSelfAttention_71262097375464
// MI455X (gfx1250) — compile-verified, matrix operand order checked by reading, not run on hardware
//
#include <hip/hip_runtime.h>

typedef __attribute__((ext_vector_type(16))) __bf16 bf16x16;
typedef __attribute__((ext_vector_type(8)))  float  v8f;

#define WMMA_BF16(A, B, C) \
  __builtin_amdgcn_wmma_f32_16x16x32_bf16(false, (A), false, (B), (short)0, (C), false, false)

union FragU2 { bf16x16 v; uint2 u[4]; };   // 8B-granular (LDS, stride not 16B-aligned)
union FragU4 { bf16x16 v; uint4 q[2]; };   // 16B-granular (global bf16, aligned)
union BPack  { __bf16 h[4]; uint2 u; };

// 16x32 bf16 fragment from bf16 global memory (row-major, stride in elements,
// rows 16B-aligned). Lane layout per CDNA5 ISA: lanes 0-15 hold K {0..7,16..23},
// lanes 16-31 hold K {8..15,24..31}; row/col = lane&15.
__device__ __forceinline__ bf16x16 load_frag_bf16g(const __bf16* __restrict__ base, int stride,
                                                   int r0, int k0, int lane) {
  const int lo = lane & 15, hi = lane >> 4;
  const __bf16* p = base + (size_t)(r0 + lo) * stride + k0 + hi * 8;
  FragU4 f;
  f.q[0] = *(const uint4*)(p);        // elements 0..7
  f.q[1] = *(const uint4*)(p + 16);   // elements 8..15
  return f.v;
}

// Same fragment from LDS bf16 (rows 8B-aligned -> uint2 pieces).
__device__ __forceinline__ bf16x16 load_frag_lds(const __bf16* base, int stride,
                                                 int r0, int k0, int lane) {
  const int lo = lane & 15, hi = lane >> 4;
  const __bf16* p = base + (r0 + lo) * stride + k0 + hi * 8;
  FragU2 f;
  f.u[0] = *(const uint2*)(p);
  f.u[1] = *(const uint2*)(p + 4);
  f.u[2] = *(const uint2*)(p + 16);
  f.u[3] = *(const uint2*)(p + 20);
  return f.v;
}

// ---------------------------------------------------------------------------
// fp32 -> bf16 pack, 4 elements/thread (n multiple of 4).
// ---------------------------------------------------------------------------
__global__ __launch_bounds__(256) void cvt_f32_to_bf16(const float* __restrict__ in,
                                                       __bf16* __restrict__ out, int n) {
  const int i = (blockIdx.x * 256 + threadIdx.x) * 4;
  if (i < n) {
    float4 v = *(const float4*)(in + i);
    BPack p;
    p.h[0] = (__bf16)v.x; p.h[1] = (__bf16)v.y;
    p.h[2] = (__bf16)v.z; p.h[3] = (__bf16)v.w;
    *(uint2*)(out + i) = p.u;
  }
}

// ---------------------------------------------------------------------------
// GEMM: C[M,N] = A[M,K] * W[N,K]^T (+ bias[N]).  bf16 in, fp32 accumulate.
// Block = 256 threads = 8 waves; wave tile 32x64 (2x4 WMMA); block tile 128x128.
// BF16_OUT: store bf16 (qkv / attn intermediates), else fp32 (+bias).
// ---------------------------------------------------------------------------
template <bool BF16_OUT>
__global__ __launch_bounds__(256) void gemm_bf16_wmma(const __bf16* __restrict__ A,
                                                      const __bf16* __restrict__ W,
                                                      const float* __restrict__ bias,
                                                      void* __restrict__ Cout,
                                                      int M, int N, int K) {
  const int lane = threadIdx.x & 31;
  const int wave = threadIdx.x >> 5;
  const int m0 = blockIdx.y * 128 + (wave & 3) * 32;   // 4 waves along M
  const int n0 = blockIdx.x * 128 + (wave >> 2) * 64;  // 2 waves along N

  v8f acc[2][4] = {{{}, {}, {}, {}}, {{}, {}, {}, {}}};

  for (int k = 0; k < K; k += 32) {
    if (k + 64 < K) {  // prefetch next-next K-block rows (global_prefetch_b8)
      __builtin_prefetch(A + (size_t)(m0 + (lane & 15)) * K + k + 64, 0, 3);
      __builtin_prefetch(W + (size_t)(n0 + (lane & 15)) * K + k + 64, 0, 3);
    }
    bf16x16 a0 = load_frag_bf16g(A, K, m0,      k, lane);
    bf16x16 a1 = load_frag_bf16g(A, K, m0 + 16, k, lane);
    bf16x16 b0 = load_frag_bf16g(W, K, n0,      k, lane);
    bf16x16 b1 = load_frag_bf16g(W, K, n0 + 16, k, lane);
    bf16x16 b2 = load_frag_bf16g(W, K, n0 + 32, k, lane);
    bf16x16 b3 = load_frag_bf16g(W, K, n0 + 48, k, lane);
    acc[0][0] = WMMA_BF16(a0, b0, acc[0][0]);
    acc[0][1] = WMMA_BF16(a0, b1, acc[0][1]);
    acc[0][2] = WMMA_BF16(a0, b2, acc[0][2]);
    acc[0][3] = WMMA_BF16(a0, b3, acc[0][3]);
    acc[1][0] = WMMA_BF16(a1, b0, acc[1][0]);
    acc[1][1] = WMMA_BF16(a1, b1, acc[1][1]);
    acc[1][2] = WMMA_BF16(a1, b2, acc[1][2]);
    acc[1][3] = WMMA_BF16(a1, b3, acc[1][3]);
  }

  const int lo = lane & 15, hi = lane >> 4;
#pragma unroll
  for (int mi = 0; mi < 2; ++mi) {
#pragma unroll
    for (int ni = 0; ni < 4; ++ni) {
      const int n = n0 + ni * 16 + lo;
      const float bn = (!BF16_OUT && bias) ? bias[n] : 0.0f;
#pragma unroll
      for (int r = 0; r < 8; ++r) {
        const int m = m0 + mi * 16 + r + 8 * hi;
        if (BF16_OUT) {
          ((__bf16*)Cout)[(size_t)m * N + n] = (__bf16)acc[mi][ni][r];
        } else {
          ((float*)Cout)[(size_t)m * N + n] = acc[mi][ni][r] + bn;
        }
      }
    }
  }
}

// ---------------------------------------------------------------------------
// Flash attention (causal). qkv fused bf16 [B*S, 3072], out bf16 [B*S, 1024].
// Grid: (S/128, B*H). Block = 256 = 8 waves; wave handles 16 query rows.
// K tile staged by ASYNCcnt DMA (global_load_async_to_lds_b128); V transposed
// manually during staging; softmax in base-2.
// ---------------------------------------------------------------------------
__global__ __launch_bounds__(256) void flash_attn_wmma(const __bf16* __restrict__ qkv,
                                                       __bf16* __restrict__ out) {
  constexpr int S = 2048, LD = 3072, DM = 1024;
  constexpr float SCALE_LOG2E = 0.125f * 1.44269504088896f;  // 1/sqrt(64) * log2(e)
  __shared__ __bf16 sK[32 * 88];        // K tile, row=key, col=depth (stride 88: 16B rows)
  __shared__ __bf16 sVt[64 * 40];       // V tile transposed, row=depth, col=key (pad 8)
  __shared__ __bf16 sP[8 * 16 * 40];    // per-wave P scratch (16 rows x 32 cols, pad 8)

  const int lane = threadIdx.x & 31;
  const int wave = threadIdx.x >> 5;
  const int lo = lane & 15, hi = lane >> 4;
  const int q0 = blockIdx.x * 128;
  const int b = blockIdx.y >> 4, h = blockIdx.y & 15;
  const __bf16* base = qkv + (size_t)b * S * LD + h * 64;   // Q channel base
  const int qrow = q0 + wave * 16;

  // Q fragments (depth 0..31 and 32..63), loaded once, bf16 direct.
  bf16x16 qa0 = load_frag_bf16g(base, LD, qrow, 0,  lane);
  bf16x16 qa1 = load_frag_bf16g(base, LD, qrow, 32, lane);

  float mrun[8], lrun[8];
  v8f acc[4] = {{}, {}, {}, {}};
#pragma unroll
  for (int r = 0; r < 8; ++r) { mrun[r] = -__builtin_inff(); lrun[r] = 0.0f; }

  const __bf16* kbase = base + 1024;
  const __bf16* vbase = base + 2048;
  __bf16* pw = sP + wave * 16 * 40;
  const int kend = q0 + 128;

  // Per-thread async K-chunk mapping: 256 chunks of 16B cover the 32x64 tile.
  const int krow = (int)threadIdx.x >> 3;          // key 0..31
  const int kcol = ((int)threadIdx.x & 7) * 8;     // depth 0..56 (16B aligned)
  const unsigned sk_dst = (unsigned)(size_t)&sK[krow * 88 + kcol];

  for (int kb = 0; kb < kend; kb += 32) {
    // --- K tile: async DMA global -> LDS, one b128 per thread (ASYNCcnt). ---
    {
      const __bf16* gsrc = kbase + (size_t)(kb + krow) * LD + kcol;
      asm volatile("global_load_async_to_lds_b128 %0, %1, off"
                   :: "v"(sk_dst), "v"(gsrc) : "memory");
    }
    // --- V tile: load + transpose into sVt (row=depth, col=key). ---
#pragma unroll
    for (int it = 0; it < 2; ++it) {
      const int f = ((int)threadIdx.x) * 2 + it;   // uint2 index 0..511
      const int row = f >> 4;                      // key 0..31
      const int dcol = (f & 15) << 2;              // depth 0..60
      BPack vv;
      vv.u = *(const uint2*)(vbase + (size_t)(kb + row) * LD + dcol);
      sVt[(dcol + 0) * 40 + row] = vv.h[0];
      sVt[(dcol + 1) * 40 + row] = vv.h[1];
      sVt[(dcol + 2) * 40 + row] = vv.h[2];
      sVt[(dcol + 3) * 40 + row] = vv.h[3];
    }
    asm volatile("s_wait_asynccnt 0x0" ::: "memory");
    __syncthreads();

    if (kb <= qrow + 15) {   // wave-uniform: skip fully-masked key blocks
      v8f s0 = {}, s1 = {};
      {
        bf16x16 k00 = load_frag_lds(sK, 88, 0,  0,  lane);
        bf16x16 k01 = load_frag_lds(sK, 88, 0,  32, lane);
        s0 = WMMA_BF16(qa0, k00, s0);
        s0 = WMMA_BF16(qa1, k01, s0);
        bf16x16 k10 = load_frag_lds(sK, 88, 16, 0,  lane);
        bf16x16 k11 = load_frag_lds(sK, 88, 16, 32, lane);
        s1 = WMMA_BF16(qa0, k10, s1);
        s1 = WMMA_BF16(qa1, k11, s1);
      }
      // Scale (base-2 domain) + causal mask (C layout: col = lo, row = r + 8*hi).
#pragma unroll
      for (int r = 0; r < 8; ++r) {
        const int rowg = qrow + r + 8 * hi;
        float v0 = s0[r] * SCALE_LOG2E;
        float v1 = s1[r] * SCALE_LOG2E;
        if (kb + lo > rowg)      v0 = -__builtin_inff();
        if (kb + 16 + lo > rowg) v1 = -__builtin_inff();
        s0[r] = v0; s1[r] = v1;
      }
      // Online softmax (exp2) per row; rows live in fixed 16-lane halves.
#pragma unroll
      for (int r = 0; r < 8; ++r) {
        float t = fmaxf(s0[r], s1[r]);
        for (int m = 8; m >= 1; m >>= 1) t = fmaxf(t, __shfl_xor(t, m, 32));
        const float mnew = fmaxf(mrun[r], t);
        const float scale = exp2f(mrun[r] - mnew);
        const float p0 = exp2f(s0[r] - mnew);
        const float p1 = exp2f(s1[r] - mnew);
        float rs = p0 + p1;
        for (int m = 8; m >= 1; m >>= 1) rs += __shfl_xor(rs, m, 32);
        lrun[r] = lrun[r] * scale + rs;
        mrun[r] = mnew;
        acc[0][r] *= scale; acc[1][r] *= scale;
        acc[2][r] *= scale; acc[3][r] *= scale;
        const int mrow = r + 8 * hi;
        pw[mrow * 40 + lo]      = (__bf16)p0;   // C-layout -> A-layout via LDS
        pw[mrow * 40 + 16 + lo] = (__bf16)p1;
      }
      // O += P(16x32) * V(32x64)
      bf16x16 pa = load_frag_lds(pw, 40, 0, 0, lane);
#pragma unroll
      for (int d = 0; d < 4; ++d) {
        bf16x16 vb = load_frag_lds(sVt, 40, d * 16, 0, lane);
        acc[d] = WMMA_BF16(pa, vb, acc[d]);
      }
    }
    __syncthreads();
  }

  // Normalize and write O (bf16) -> [b*S + row, h*64 + d].
#pragma unroll
  for (int r = 0; r < 8; ++r) {
    const float inv = 1.0f / lrun[r];
    const int rowg = qrow + r + 8 * hi;
    __bf16* o = out + (size_t)(b * S + rowg) * DM + h * 64;
    o[0 * 16 + lo] = (__bf16)(acc[0][r] * inv);
    o[1 * 16 + lo] = (__bf16)(acc[1][r] * inv);
    o[2 * 16 + lo] = (__bf16)(acc[2][r] * inv);
    o[3 * 16 + lo] = (__bf16)(acc[3][r] * inv);
  }
}

// ---------------------------------------------------------------------------
extern "C" void kernel_launch(void* const* d_in, const int* in_sizes, int n_in,
                              void* d_out, int out_size, void* d_ws, size_t ws_size,
                              hipStream_t stream) {
  (void)in_sizes; (void)n_in; (void)out_size; (void)ws_size;
  const float* x    = (const float*)d_in[0];   // [2,2048,1024]
  const float* Wqkv = (const float*)d_in[1];   // [3072,1024]
  const float* Wout = (const float*)d_in[2];   // [1024,1024]
  const float* bout = (const float*)d_in[3];   // [1024]
  float* outp = (float*)d_out;                 // [2,2048,1024] fp32

  constexpr int M = 2 * 2048;                  // B*S
  constexpr int NX  = M * 1024;                // x elems
  constexpr int NWQ = 3072 * 1024;
  constexpr int NWO = 1024 * 1024;

  __bf16* xb    = (__bf16*)d_ws;               //  8 MB
  __bf16* wqb   = xb + NX;                     //  6 MB
  __bf16* wob   = wqb + NWQ;                   //  2 MB
  __bf16* qkvb  = wob + NWO;                   // 24 MB
  __bf16* attnb = qkvb + (size_t)M * 3072;     //  8 MB

  // 0) One-time fp32 -> bf16 conversions.
  cvt_f32_to_bf16<<<NX  / 1024, 256, 0, stream>>>(x,    xb,  NX);
  cvt_f32_to_bf16<<<NWQ / 1024, 256, 0, stream>>>(Wqkv, wqb, NWQ);
  cvt_f32_to_bf16<<<NWO / 1024, 256, 0, stream>>>(Wout, wob, NWO);

  // 1) QKV projection: qkv = x @ Wqkv^T   (bf16 out for attention)
  gemm_bf16_wmma<true><<<dim3(3072 / 128, M / 128), 256, 0, stream>>>(
      xb, wqb, nullptr, qkvb, M, 3072, 1024);

  // 2) Causal flash attention per (b, h), 128 query rows per block.
  flash_attn_wmma<<<dim3(2048 / 128, 2 * 16), 256, 0, stream>>>(qkvb, attnb);

  // 3) Output projection: out = attn @ Wout^T + b_out   (fp32 out)
  gemm_bf16_wmma<false><<<dim3(1024 / 128, M / 128), 256, 0, stream>>>(
      attnb, wob, bout, outp, M, 1024, 1024);
}
